// _Expert_11871289606694
// MI455X (gfx1250) — matrix operand order repaired and checked by reading, not compiled, not run on hardware
//
#include <hip/hip_runtime.h>
#include <hip/hip_bf16.h>

// MoE grouped GEMM: out[T, N] = inp[T, K] @ weight[e]^T + bias[e]
//   E=16 experts, K=D_IN=2048, N=D_OUT=8192, T=8192 tokens, 512 tokens/expert.
// bf16x3 split-precision WMMA GEMM (ah*bh + ah*bl + al*bh) on v_wmma_f32_16x16x32_bf16.

#define NUM_E   16
#define DIM_K   2048
#define DIM_N   8192
#define NUM_T   8192
#define BM      128
#define BN      128
#define BK      32
#define LDS_STR 40   // padded row stride (elements) -> 80B rows, conflict-free frag reads

typedef __attribute__((ext_vector_type(16))) __bf16 v16bf;
typedef __attribute__((ext_vector_type(8)))  __bf16 v8bf;
typedef __attribute__((ext_vector_type(4)))  __bf16 v4bf;
typedef __attribute__((ext_vector_type(8)))  float  v8f;

union ABfrag { v16bf v; v8bf h[2]; };

struct HL { __bf16 hi, lo; };
__device__ __forceinline__ HL split2(float x) {
    HL r;
    __bf16 h = (__bf16)x;
    r.hi = h;
    r.lo = (__bf16)(x - (float)h);
    return r;
}

__global__ __launch_bounds__(256) void moe_grouped_gemm_bf16x3(
    const float* __restrict__ inp,      // [T, K]
    const float* __restrict__ weight,   // [E, N, K]
    const float* __restrict__ bias,     // [E, N]
    const int*   __restrict__ counts,   // [E]
    float*       __restrict__ out)      // [T, N]
{
    __shared__ __bf16 sAh[BM * LDS_STR];
    __shared__ __bf16 sAl[BM * LDS_STR];
    __shared__ __bf16 sBh[BN * LDS_STR];
    __shared__ __bf16 sBl[BN * LDS_STR];

    const int tid   = threadIdx.x;
    const int lane  = tid & 31;
    const int wave  = tid >> 5;
    const int wm    = wave >> 2;        // 0..1  -> 64-row slab
    const int wn    = wave & 3;         // 0..3  -> 32-col slab
    const int l16   = lane & 15;
    const int lhalf = lane >> 4;        // 0/1

    const int n0 = blockIdx.x * BN;     // output-column tile
    const int m0 = blockIdx.y * BM;     // token-row tile

    // expert for this M tile (tiles never straddle experts: counts are 512 here)
    int e = 0;
    {
        int pre = 0;
        for (int i = 0; i < NUM_E; ++i) {
            int c = counts[i];
            if (m0 >= pre + c) { pre += c; } else { e = i; break; }
        }
    }

    // staging map: 256 threads cover 128 rows x 8 float4-chunks, 4 sweeps of 32 rows
    const int srow = tid >> 3;          // 0..31
    const int skc  = tid & 7;           // 0..7 -> float4 chunk within BK

    const float* gA = inp    + (size_t)(m0)*DIM_K + skc * 4;
    const float* gB = weight + ((size_t)e * DIM_N + n0) * DIM_K + skc * 4;

    v8f acc[4][2];
    #pragma unroll
    for (int mf = 0; mf < 4; ++mf)
        #pragma unroll
        for (int nf = 0; nf < 2; ++nf)
            acc[mf][nf] = (v8f)(0.0f);

    for (int k0 = 0; k0 < DIM_K; k0 += BK) {
        // -------- stage global -> registers (fp32) --------
        float4 ra[4], rb[4];
        #pragma unroll
        for (int it = 0; it < 4; ++it) {
            int row = srow + it * 32;
            ra[it] = *(const float4*)(gA + (size_t)row * DIM_K + k0);
            rb[it] = *(const float4*)(gB + (size_t)row * DIM_K + k0);
        }

        __syncthreads();   // previous iteration's fragment reads complete

        // -------- split fp32 -> (hi, lo) bf16, write LDS --------
        #pragma unroll
        for (int it = 0; it < 4; ++it) {
            int row = srow + it * 32;
            int off = row * LDS_STR + skc * 4;
            v4bf ah4, al4, bh4, bl4;
            HL t;
            t = split2(ra[it].x); ah4[0] = t.hi; al4[0] = t.lo;
            t = split2(ra[it].y); ah4[1] = t.hi; al4[1] = t.lo;
            t = split2(ra[it].z); ah4[2] = t.hi; al4[2] = t.lo;
            t = split2(ra[it].w); ah4[3] = t.hi; al4[3] = t.lo;
            t = split2(rb[it].x); bh4[0] = t.hi; bl4[0] = t.lo;
            t = split2(rb[it].y); bh4[1] = t.hi; bl4[1] = t.lo;
            t = split2(rb[it].z); bh4[2] = t.hi; bl4[2] = t.lo;
            t = split2(rb[it].w); bh4[3] = t.hi; bl4[3] = t.lo;
            *(v4bf*)(&sAh[off]) = ah4;
            *(v4bf*)(&sAl[off]) = al4;
            *(v4bf*)(&sBh[off]) = bh4;
            *(v4bf*)(&sBl[off]) = bl4;
        }

        __syncthreads();   // tiles visible

        // -------- B fragments (lane = N col; per-lane contiguous K run) --------
        ABfrag bh[2], bl[2];
        #pragma unroll
        for (int nf = 0; nf < 2; ++nf) {
            const __bf16* brh = &sBh[(wn * 32 + nf * 16 + l16) * LDS_STR];
            const __bf16* brl = &sBl[(wn * 32 + nf * 16 + l16) * LDS_STR];
            int kb = lhalf * 16;    // lanes 16-31 hold K=16..31
            bh[nf].h[0] = *(const v8bf*)(brh + kb);
            bh[nf].h[1] = *(const v8bf*)(brh + kb + 8);
            bl[nf].h[0] = *(const v8bf*)(brl + kb);
            bl[nf].h[1] = *(const v8bf*)(brl + kb + 8);
        }

        // -------- A fragments + WMMA --------
        #pragma unroll
        for (int mf = 0; mf < 4; ++mf) {
            const __bf16* arh = &sAh[(wm * 64 + mf * 16 + l16) * LDS_STR];
            const __bf16* arl = &sAl[(wm * 64 + mf * 16 + l16) * LDS_STR];
            int kb = lhalf * 16;    // lanes 0-15: K 0..15; lanes 16-31: K 16..31
            ABfrag ah, al;
            ah.h[0] = *(const v8bf*)(arh + kb);
            ah.h[1] = *(const v8bf*)(arh + kb + 8);
            al.h[0] = *(const v8bf*)(arl + kb);
            al.h[1] = *(const v8bf*)(arl + kb + 8);

            #pragma unroll
            for (int nf = 0; nf < 2; ++nf) {
                v8f c = acc[mf][nf];
                c = __builtin_amdgcn_wmma_f32_16x16x32_bf16(
                        false, ah.v, false, bh[nf].v, (short)0, c, false, false);
                c = __builtin_amdgcn_wmma_f32_16x16x32_bf16(
                        false, ah.v, false, bl[nf].v, (short)0, c, false, false);
                c = __builtin_amdgcn_wmma_f32_16x16x32_bf16(
                        false, al.v, false, bh[nf].v, (short)0, c, false, false);
                acc[mf][nf] = c;
            }
        }
    }

    // -------- epilogue: bias + store --------
    // C/D layout: lane 0-15 -> N=lane, VGPR r -> M=r; lanes 16-31 -> M=r+8
    #pragma unroll
    for (int nf = 0; nf < 2; ++nf) {
        int col = n0 + wn * 32 + nf * 16 + l16;
        float bv = bias[(size_t)e * DIM_N + col];
        #pragma unroll
        for (int mf = 0; mf < 4; ++mf) {
            int mbase = m0 + wm * 64 + mf * 16 + lhalf * 8;
            #pragma unroll
            for (int r = 0; r < 8; ++r) {
                out[(size_t)(mbase + r) * DIM_N + col] = acc[mf][nf][r] + bv;
            }
        }
    }
}

extern "C" void kernel_launch(void* const* d_in, const int* in_sizes, int n_in,
                              void* d_out, int out_size, void* d_ws, size_t ws_size,
                              hipStream_t stream) {
    (void)in_sizes; (void)n_in; (void)out_size; (void)d_ws; (void)ws_size;
    const float* inp    = (const float*)d_in[0];
    const float* weight = (const float*)d_in[1];
    const float* bias   = (const float*)d_in[2];
    const int*   counts = (const int*)d_in[3];
    float* out = (float*)d_out;

    dim3 grid(DIM_N / BN, NUM_T / BM);   // 64 x 64 tiles
    dim3 block(256);
    moe_grouped_gemm_bf16x3<<<grid, block, 0, stream>>>(inp, weight, bias, counts, out);
}
